// ConvAttention_11811160064402
// MI455X (gfx1250) — matrix operand order repaired and checked by reading, not compiled, not run on hardware
//
#include <hip/hip_runtime.h>

// ---------------------------------------------------------------------------
// ConvAttention for MI455X (gfx1250, wave32, WMMA bf16 16x16x32)
// B=2, L=W=64, N=4096, C=128, H=4, DH=32
// ---------------------------------------------------------------------------

#define B_   2
#define L_   64
#define W_   64
#define N_   4096
#define C_   128
#define H_   4
#define DH_  32
#define SCALE_ 0.17677669529663687f   // 32^-0.5
#define BNEPS_ 1e-3f

typedef __attribute__((ext_vector_type(16))) __bf16       bf16x16;
typedef __attribute__((ext_vector_type(8)))  float        f32x8;
typedef __attribute__((ext_vector_type(4)))  unsigned int u32x4;

union FragU { bf16x16 v; u32x4 q[2]; };

// Load a 16-element bf16 fragment as two 16-byte chunks (A-frag: chunks are
// 16 elements apart in the source row; B-frag: likewise, p1 = p0 + 16).
__device__ __forceinline__ bf16x16 load_frag_pair(const __bf16* p0, const __bf16* p1) {
  FragU f;
  f.q[0] = *(const u32x4*)p0;
  f.q[1] = *(const u32x4*)p1;
  return f.v;
}

__device__ __forceinline__ f32x8 wmma_bf16(bf16x16 a, bf16x16 b, f32x8 c) {
  // D = A(16x32) * B(32x16) + C, f32 accumulate
  return __builtin_amdgcn_wmma_f32_16x16x32_bf16(false, a, false, b, (short)0, c,
                                                 false, false);
}

// ---------------------------------------------------------------------------
// Kernel 0: fold BN into pointwise weights / bias, transpose weights to
// [c_out][c_in] bf16 so WMMA B-fragments are contiguous loads.
//   y = a_c * dw_c + beta_c,  a = g*rsqrt(var+eps), beta = a*(dwb-mu)+b
//   z_f = sum_c dw_c * (a_c*pwk[c][f]) + (sum_c beta_c*pwk[c][f] + pwb[f])
// ---------------------------------------------------------------------------
struct PrepArgs {
  const float *g[3], *b[3], *mu[3], *var[3], *dwb[3], *pwk[3], *pwb[3];
  const float *out_w;
  __bf16 *Wt[3];      // [128][128] bf16, c_out-major
  float  *bias2[3];   // [128] f32
  __bf16 *WoT;        // out_w transposed, [f][c] bf16
};

__global__ void prep_kernel(PrepArgs A) {
  int f = threadIdx.x;
  for (int m = 0; m < 3; ++m) {
    float bias = A.pwb[m][f];
    for (int c = 0; c < C_; ++c) {
      float a    = A.g[m][c] * rsqrtf(A.var[m][c] + BNEPS_);
      float beta = a * (A.dwb[m][c] - A.mu[m][c]) + A.b[m][c];
      float w    = A.pwk[m][c * C_ + f];
      A.Wt[m][f * C_ + c] = (__bf16)(a * w);
      bias += beta * w;
    }
    A.bias2[m][f] = bias;
  }
  for (int c = 0; c < C_; ++c)
    A.WoT[f * C_ + c] = (__bf16)A.out_w[c * C_ + f];
}

// ---------------------------------------------------------------------------
// Kernel 1: separable conv -> Q/K (row-major [b,h,n,32] bf16), V transposed
// ([b,h,32,n] bf16).  One block = one (matrix m, batch b, image row l):
// 64 pixels x 128 channels.  Depthwise in fp32 -> bf16 LDS -> WMMA pointwise.
// Epilogue is branchless: Q/K and V^T share base offset bh*N*DH and differ
// only in (pixel stride, channel stride), chosen uniformly per block.
// ---------------------------------------------------------------------------
struct SepArgs {
  const float*  x;
  const float*  dwk[3];     // (3,3,1,128)
  const __bf16* Wt[3];
  const float*  bias2[3];
  __bf16 *Q, *K, *V;        // V is DH-major (transposed)
};

__global__ void sepconv_kernel(SepArgs A) {
  __shared__ __bf16 ybuf[64 * C_];   // 16 KB
  int t = blockIdx.x;
  int m = t / (B_ * L_); t -= m * (B_ * L_);
  int b    = t / L_;
  int lrow = t % L_;
  int tid  = threadIdx.x;

  // Depthwise 3x3: thread handles channel tid for all 64 pixels of row lrow.
  float kw[9];
#pragma unroll
  for (int j = 0; j < 9; ++j) kw[j] = A.dwk[m][j * C_ + tid];
  const float* __restrict__ xb = A.x + (size_t)b * N_ * C_;
  for (int w = 0; w < W_; ++w) {
    float acc = 0.f;
#pragma unroll
    for (int dy = -1; dy <= 1; ++dy) {
      int ll = lrow + dy;
      if (ll < 0 || ll >= L_) continue;
#pragma unroll
      for (int dx = -1; dx <= 1; ++dx) {
        int ww = w + dx;
        if (ww < 0 || ww >= W_) continue;
        acc += xb[(size_t)(ll * W_ + ww) * C_ + tid] * kw[(dy + 1) * 3 + (dx + 1)];
      }
    }
    ybuf[w * C_ + tid] = (__bf16)acc;
  }
  __syncthreads();

  // Pointwise 1x1 as WMMA: wave wv owns pixels wv*16..+15 (A rows).
  int lane = tid & 31, wv = tid >> 5;
  int lo = lane & 15, hi = lane >> 4;

  bf16x16 afrag[4];
  {
    int row = wv * 16 + lo;
    const __bf16* rb = &ybuf[row * C_ + hi * 8];
#pragma unroll
    for (int kc = 0; kc < 4; ++kc)
      afrag[kc] = load_frag_pair(rb + kc * 32, rb + kc * 32 + 16);
  }

  // Uniform destination selection (no per-element branches):
  //   Q/K: addr = base + bh*N*DH + p*DH + dd*1
  //   V^T: addr = base + bh*N*DH + p*1  + dd*N
  __bf16* __restrict__ dst = (m == 0) ? A.Q : (m == 1) ? A.K : A.V;
  int pstride = (m == 2) ? 1 : DH_;
  int dstride = (m == 2) ? N_ : 1;

  const __bf16* __restrict__ Wt    = A.Wt[m];
  const float*  __restrict__ bias2 = A.bias2[m];
#pragma unroll
  for (int n = 0; n < 8; ++n) {
    int colc = n * 16 + lo;
    const __bf16* wb = Wt + (size_t)colc * C_ + hi * 8;
    f32x8 acc = {};
#pragma unroll
    for (int kc = 0; kc < 4; ++kc)
      acc = wmma_bf16(afrag[kc], load_frag_pair(wb + kc * 32, wb + kc * 32 + 16), acc);

    float bias = bias2[colc];
    int hh = colc >> 5, dd = colc & 31;
    size_t base = (size_t)(b * H_ + hh) * (N_ * DH_) + (size_t)dd * dstride;
#pragma unroll
    for (int r = 0; r < 8; ++r) {
      int p = lrow * W_ + wv * 16 + r + hi * 8;      // D layout: row = r + 8*hi
      dst[base + (size_t)p * pstride] = (__bf16)(acc[r] + bias);
    }
  }
}

// ---------------------------------------------------------------------------
// Kernel 2: flash attention.  One block = (b, h, 64 queries); each wave owns
// 16 query rows.  S^T = K_chunk @ Q^T  (query = lane, key = VGPR) so online
// softmax is per-lane + one shfl_xor(16), and the P fragment is a lane-local
// repack of the S^T accumulators.  Next K/V chunk is prefetched
// (global_prefetch_b8) to hide L2 hit latency behind the 4 WMMAs + softmax.
// ---------------------------------------------------------------------------
struct AttnArgs { const __bf16 *Q, *K, *V; __bf16* att; };

__global__ void attn_kernel(AttnArgs A) {
  int t  = blockIdx.x;
  int b  = t / (H_ * (N_ / 64));
  int h  = (t / (N_ / 64)) % H_;
  int q0 = (t % (N_ / 64)) * 64;
  int tid = threadIdx.x;
  int lane = tid & 31, wv = tid >> 5;
  int lo = lane & 15, hi = lane >> 4;

  size_t bh = (size_t)(b * H_ + h);
  const __bf16* __restrict__ Qb = A.Q + bh * N_ * DH_;
  const __bf16* __restrict__ Kb = A.K + bh * N_ * DH_;
  const __bf16* __restrict__ Vb = A.V + bh * DH_ * N_;

  // B-fragment of Q^T: col = query (lane), K dim = d (contiguous in memory).
  int qrow = q0 + wv * 16 + lo;
  const __bf16* qp = Qb + (size_t)qrow * DH_ + hi * 8;
  bf16x16 qfrag = load_frag_pair(qp, qp + 16);

  float mrow = -3.0e38f, lsum = 0.f;
  f32x8 o0 = {}, o1 = {};

  for (int kb = 0; kb < N_; kb += 32) {
    // A-fragments of two 16-key tiles of K (row = key, K dim = d).
    const __bf16* k0p = Kb + (size_t)(kb + lo) * DH_ + hi * 8;
    const __bf16* k1p = Kb + (size_t)(kb + 16 + lo) * DH_ + hi * 8;
    bf16x16 kf0 = load_frag_pair(k0p, k0p + 16);
    bf16x16 kf1 = load_frag_pair(k1p, k1p + 16);

    // Prefetch next chunk's K and V cachelines (global_prefetch_b8).
    if (kb + 32 < N_) {
      __builtin_prefetch(Kb + (size_t)(kb + 32 + lo) * DH_, 0, 3);
      __builtin_prefetch(Vb + (size_t)lo * N_ + kb + 32, 0, 3);
      __builtin_prefetch(Vb + (size_t)(16 + lo) * N_ + kb + 32, 0, 3);
    }

    f32x8 z = {};
    f32x8 s0 = wmma_bf16(kf0, qfrag, z);
    f32x8 s1 = wmma_bf16(kf1, qfrag, z);

    float cmax = -3.0e38f;
#pragma unroll
    for (int r = 0; r < 8; ++r) {
      s0[r] *= SCALE_; s1[r] *= SCALE_;
      cmax = fmaxf(cmax, fmaxf(s0[r], s1[r]));
    }
    cmax = fmaxf(cmax, __shfl_xor(cmax, 16, 32));   // combine key halves
    float mnew  = fmaxf(mrow, cmax);
    float alpha = __expf(mrow - mnew);
    mrow = mnew;

    float csum = 0.f;
    bf16x16 pf;
#pragma unroll
    for (int r = 0; r < 8; ++r) {                   // P = exp(S - m); repack
      float p0v = __expf(s0[r] - mnew);             // S^T D-frag -> A-frag,
      float p1v = __expf(s1[r] - mnew);             // purely lane-local.
      csum += p0v + p1v;
      pf[r]     = (__bf16)p0v;
      pf[8 + r] = (__bf16)p1v;
    }
    csum += __shfl_xor(csum, 16, 32);
    lsum = lsum * alpha + csum;

#pragma unroll
    for (int r = 0; r < 8; ++r) {                   // rescale accumulators
      float ar = __shfl(alpha, r + hi * 8, 32);     // alpha of query row
      o0[r] *= ar; o1[r] *= ar;
    }

    // B-fragments of V chunk (col = d, K dim = key; V stored DH-major).
    const __bf16* v0p = Vb + (size_t)lo * N_ + kb + hi * 8;
    const __bf16* v1p = Vb + (size_t)(16 + lo) * N_ + kb + hi * 8;
    o0 = wmma_bf16(pf, load_frag_pair(v0p, v0p + 16), o0);
    o1 = wmma_bf16(pf, load_frag_pair(v1p, v1p + 16), o1);
  }

  float inv = 1.f / lsum;
#pragma unroll
  for (int r = 0; r < 8; ++r) {
    float sc = __shfl(inv, r + hi * 8, 32);
    int qr = q0 + wv * 16 + r + hi * 8;
    size_t base = ((size_t)b * N_ + qr) * C_ + h * DH_;
    A.att[base + lo]      = (__bf16)(o0[r] * sc);
    A.att[base + 16 + lo] = (__bf16)(o1[r] * sc);
  }
}

// ---------------------------------------------------------------------------
// Kernel 3: output projection out = att @ out_w + out_b, fp32 result.
// ---------------------------------------------------------------------------
struct ProjArgs { const __bf16* att; const __bf16* WoT; const float* out_b; float* out; };

__global__ void proj_kernel(ProjArgs A) {
  int row0 = blockIdx.x * 64;
  int tid = threadIdx.x, lane = tid & 31, wv = tid >> 5;
  int lo = lane & 15, hi = lane >> 4;

  int row = row0 + wv * 16 + lo;
  const __bf16* __restrict__ ab = A.att + (size_t)row * C_ + hi * 8;
  bf16x16 afrag[4];
#pragma unroll
  for (int kc = 0; kc < 4; ++kc)
    afrag[kc] = load_frag_pair(ab + kc * 32, ab + kc * 32 + 16);

#pragma unroll
  for (int n = 0; n < 8; ++n) {
    int colc = n * 16 + lo;
    const __bf16* wb = A.WoT + (size_t)colc * C_ + hi * 8;
    f32x8 acc = {};
#pragma unroll
    for (int kc = 0; kc < 4; ++kc)
      acc = wmma_bf16(afrag[kc], load_frag_pair(wb + kc * 32, wb + kc * 32 + 16), acc);

    float bias = A.out_b[colc];
#pragma unroll
    for (int r = 0; r < 8; ++r) {
      int rr = row0 + wv * 16 + r + hi * 8;
      A.out[(size_t)rr * C_ + colc] = acc[r] + bias;
    }
  }
}

// ---------------------------------------------------------------------------
// Host launcher.  Inputs (setup_inputs order): x, then per q/k/v:
// dwk,dwb,g,b,mu,var,pwk,pwb, then out_w, out_b.
// Workspace layout (~9 MB): folded weights/biases, Q, K, V^T, att.
// ---------------------------------------------------------------------------
extern "C" void kernel_launch(void* const* d_in, const int* in_sizes, int n_in,
                              void* d_out, int out_size, void* d_ws, size_t ws_size,
                              hipStream_t stream) {
  (void)in_sizes; (void)n_in; (void)out_size; (void)ws_size;

  const float* x = (const float*)d_in[0];
  const float* P[3][8];
  for (int m = 0; m < 3; ++m)
    for (int j = 0; j < 8; ++j)
      P[m][j] = (const float*)d_in[1 + m * 8 + j];
  const float* out_w = (const float*)d_in[25];
  const float* out_b = (const float*)d_in[26];

  char* ws = (char*)d_ws;
  __bf16* Wt[3]    = {(__bf16*)(ws + 0), (__bf16*)(ws + 32768), (__bf16*)(ws + 65536)};
  __bf16* WoT      = (__bf16*)(ws + 98304);
  float*  bias2[3] = {(float*)(ws + 131072), (float*)(ws + 131584), (float*)(ws + 132096)};
  __bf16* Qb  = (__bf16*)(ws + (1ull << 20));
  __bf16* Kb  = (__bf16*)(ws + (3ull << 20));
  __bf16* Vb  = (__bf16*)(ws + (5ull << 20));
  __bf16* att = (__bf16*)(ws + (7ull << 20));

  PrepArgs pa;
  for (int m = 0; m < 3; ++m) {
    pa.dwb[m] = P[m][1]; pa.g[m]  = P[m][2]; pa.b[m]   = P[m][3];
    pa.mu[m]  = P[m][4]; pa.var[m] = P[m][5];
    pa.pwk[m] = P[m][6]; pa.pwb[m] = P[m][7];
    pa.Wt[m] = Wt[m]; pa.bias2[m] = bias2[m];
  }
  pa.out_w = out_w; pa.WoT = WoT;
  prep_kernel<<<1, 128, 0, stream>>>(pa);

  SepArgs sa; sa.x = x;
  for (int m = 0; m < 3; ++m) { sa.dwk[m] = P[m][0]; sa.Wt[m] = Wt[m]; sa.bias2[m] = bias2[m]; }
  sa.Q = Qb; sa.K = Kb; sa.V = Vb;
  sepconv_kernel<<<3 * B_ * L_, 128, 0, stream>>>(sa);

  AttnArgs aa; aa.Q = Qb; aa.K = Kb; aa.V = Vb; aa.att = att;
  attn_kernel<<<B_ * H_ * (N_ / 64), 128, 0, stream>>>(aa);

  ProjArgs pra; pra.att = att; pra.WoT = WoT; pra.out_b = out_b; pra.out = (float*)d_out;
  proj_kernel<<<(B_ * N_) / 64, 128, 0, stream>>>(pra);
}
